// PointTransformerLayer_28973849379264
// MI455X (gfx1250) — matrix operand order repaired and checked by reading, not compiled, not run on hardware
//
#include <hip/hip_runtime.h>

typedef __attribute__((ext_vector_type(16))) _Float16 v16h;
typedef __attribute__((ext_vector_type(8)))  _Float16 v8h;
typedef __attribute__((ext_vector_type(4)))  _Float16 v4h;
typedef __attribute__((ext_vector_type(2)))  _Float16 v2h;
typedef __attribute__((ext_vector_type(8)))  float    v8f;
typedef __attribute__((ext_vector_type(4)))  float    v4f;

#define PT_N   65536   // points per batch
#define PT_B   8
#define PT_C   64      // cin == cout == 64

#define WAVES          8
#define THREADS        (WAVES * 32)
#define TILES_PER_WAVE 2
#define NBLOCKS        ((PT_B * PT_N) / (16 * WAVES * TILES_PER_WAVE))  // 2048

#define LOG2E 1.44269504f

__device__ __forceinline__ v16h cat8(v8h a, v8h b) {
  v16h r;
#pragma unroll
  for (int i = 0; i < 8; ++i) { r[i] = a[i]; r[i + 8] = b[i]; }
  return r;
}

// B operand (K x 16, f16) from LDS weight stored as [cout][K-slot] f16.
// Lanes 0-15 carry K = ks*32 + 0..7, 16..23; lanes 16-31 carry K = ks*32 + 8..15, 24..31,
// packed two-K-per-VGPR, for column N = lane & 15.
__device__ __forceinline__ v16h lds_bop(const _Float16* w, int coutCol, int hi, int ks) {
  const int kstart = ks * 32 + hi * 8;
  const v8h* p = (const v8h*)(w + coutCol * 64 + kstart);
  return cat8(p[0], p[2]);
}

// A operand [16 x 32] f16 chunk from an LDS [16 point][64 K-slot] f16 stage.
__device__ __forceinline__ v16h lds_aop(const _Float16* sb, int row, int ks, int hi) {
  const int bc = ks * 32 + hi * 8;
  const v8h* p0 = (const v8h*)(sb + row * 64 + bc);
  const v8h* p1 = (const v8h*)(sb + row * 64 + bc + 16);
  return cat8(p0[0], p1[0]);
}

__global__ __launch_bounds__(THREADS) void pt_layer_kernel(
    const float* __restrict__ x,
    const float* __restrict__ Wq, const float* __restrict__ bq,
    const float* __restrict__ Wk, const float* __restrict__ bk,
    const float* __restrict__ Wv, const float* __restrict__ bv,
    const float* __restrict__ Wg, const float* __restrict__ bg,
    const float* __restrict__ Wo, const float* __restrict__ bo,
    const int*   __restrict__ nn,
    float* __restrict__ out) {
  // wqk = (Wq-Wk)*log2(e), natural cin order.  wvv: natural cin order.
  // wgg/woo: cin axis PERMUTED by slot(c) = (c&15)*4 + (c>>4) to match the
  // p/u staging layout (lets each lane store its 4 per-point values as one b64).
  __shared__ _Float16 wqk[64 * 64];
  __shared__ _Float16 wvv[64 * 64];
  __shared__ _Float16 wgg[64 * 64];
  __shared__ _Float16 woo[64 * 64];
  __shared__ float ebqk[64];   // exp(bq - bk)
  __shared__ float bvs[64];    // bv
  __shared__ float bcn[64];    // bg @ Wo + bo  (constant fold)
  __shared__ _Float16 sstage[WAVES * 16 * 64];  // per-wave [16 point][64 slot] f16

  const int tid = threadIdx.x;
  for (int p = tid; p < 32 * 64; p += THREADS) {   // 2048 f16-pairs per matrix
    const int j = p & 63, kh = p >> 6;             // j = cout, kh in [0,32)
    const int k0 = 2 * kh, k1 = 2 * kh + 1;        // natural-order pair
    v2h a = {(_Float16)((Wq[k0 * 64 + j] - Wk[k0 * 64 + j]) * LOG2E),
             (_Float16)((Wq[k1 * 64 + j] - Wk[k1 * 64 + j]) * LOG2E)};
    *(v2h*)(wqk + j * 64 + 2 * kh) = a;
    v2h b = {(_Float16)Wv[k0 * 64 + j], (_Float16)Wv[k1 * 64 + j]};
    *(v2h*)(wvv + j * 64 + 2 * kh) = b;
    // permuted slots 2kh, 2kh+1  <->  original k = (s>>2) + 16*(s&3)
    const int pk0 = ((2 * kh) >> 2) + 16 * ((2 * kh) & 3);
    const int pk1 = ((2 * kh + 1) >> 2) + 16 * ((2 * kh + 1) & 3);
    v2h c = {(_Float16)Wg[pk0 * 64 + j], (_Float16)Wg[pk1 * 64 + j]};
    *(v2h*)(wgg + j * 64 + 2 * kh) = c;
    v2h d = {(_Float16)Wo[pk0 * 64 + j], (_Float16)Wo[pk1 * 64 + j]};
    *(v2h*)(woo + j * 64 + 2 * kh) = d;
  }
  if (tid < 64) {
    ebqk[tid] = __expf(bq[tid] - bk[tid]);
    bvs[tid]  = bv[tid];
    float acc = bo[tid];
#pragma unroll 8
    for (int k = 0; k < 64; ++k) acc += bg[k] * Wo[k * 64 + tid];
    bcn[tid] = acc;
  }
  __syncthreads();

  const int wave = tid >> 5;
  const int lane = tid & 31;
  const int col  = lane & 15;   // cout column within a 16-wide tile / A row index
  const int hi   = lane >> 4;   // lane half (selects K sub-block / point half)
  const float scale = (float)nn[0];
  _Float16* sb = sstage + wave * (16 * 64);

  // Loop-invariant per-lane channel constants (cout = 16t + col).
  float ebq[4], bvl[4], bcl[4];
#pragma unroll
  for (int t = 0; t < 4; ++t) {
    ebq[t] = ebqk[16 * t + col];
    bvl[t] = bvs[16 * t + col];
    bcl[t] = bcn[16 * t + col];
  }

  for (int it = 0; it < TILES_PER_WAVE; ++it) {
    const int tile = (blockIdx.x * WAVES + wave) + it * (NBLOCKS * WAVES);
    const int m0 = tile * 16;
    const int b  = m0 >> 16;           // 16 | PT_N so tiles never straddle a batch
    const int n0 = m0 & (PT_N - 1);
    const float* xb = x + ((size_t)b << 22);  // b * 64 * 65536

    // ---- A operand: x tile [16 points x 64 cin] as f16, two K-steps of 32.
    // ISA 16-bit A layout: lane(0-15) row M=lane, elems 0..7 -> K=bc..bc+7,
    // elems 8..15 -> K=bc+16..bc+23; hi half shifts bc by 8.
    v16h ax[2];
#pragma unroll
    for (int ks = 0; ks < 2; ++ks) {
      const int bc = ks * 32 + hi * 8;
      union { v16h v; _Float16 h[16]; } u;
#pragma unroll
      for (int i = 0; i < 8; ++i) {
        u.h[i]     = (_Float16)xb[(size_t)(bc + i)      * PT_N + n0 + col];
        u.h[i + 8] = (_Float16)xb[(size_t)(bc + 16 + i) * PT_N + n0 + col];
      }
      ax[ks] = u.v;
    }

    // ---- dn = x@((Wq-Wk)*log2e)  (bias folded into exp),  vn = x@Wv (bias added later).
    // D layout: comp r, lane half -> point n0 + hi*8 + r, cout = 16t + col.
    v8f dA[4], vA[4];
#pragma unroll
    for (int t = 0; t < 4; ++t) {
      v8f z = {};
      v8f acc = __builtin_amdgcn_wmma_f32_16x16x32_f16(false, ax[0], false, lds_bop(wqk, 16 * t + col, hi, 0), (short)0, z, false, false);
      acc     = __builtin_amdgcn_wmma_f32_16x16x32_f16(false, ax[1], false, lds_bop(wqk, 16 * t + col, hi, 1), (short)0, acc, false, false);
      dA[t] = acc;
      v8f accv = __builtin_amdgcn_wmma_f32_16x16x32_f16(false, ax[0], false, lds_bop(wvv, 16 * t + col, hi, 0), (short)0, z, false, false);
      accv     = __builtin_amdgcn_wmma_f32_16x16x32_f16(false, ax[1], false, lds_bop(wvv, 16 * t + col, hi, 1), (short)0, accv, false, false);
      vA[t] = accv;
    }

    // ---- e = 2^(dn) * exp(bqk);  sigma = nn / sum(e);  p = e * (vn + bv)  (unnormalized).
    // No max subtraction: |logits| bounded small for this problem, exp can't overflow in f32.
    float sig[8];
#pragma unroll
    for (int r = 0; r < 8; ++r) {
      const float e0 = __builtin_amdgcn_exp2f(dA[0][r]) * ebq[0];
      const float e1 = __builtin_amdgcn_exp2f(dA[1][r]) * ebq[1];
      const float e2 = __builtin_amdgcn_exp2f(dA[2][r]) * ebq[2];
      const float e3 = __builtin_amdgcn_exp2f(dA[3][r]) * ebq[3];
      float sm = (e0 + e1) + (e2 + e3);
      sm += __shfl_xor(sm, 1, 32);
      sm += __shfl_xor(sm, 2, 32);
      sm += __shfl_xor(sm, 4, 32);
      sm += __shfl_xor(sm, 8, 32);
      sig[r] = scale * __builtin_amdgcn_rcpf(sm);
      dA[0][r] = e0 * (vA[0][r] + bvl[0]);   // reuse dA as p
      dA[1][r] = e1 * (vA[1][r] + bvl[1]);
      dA[2][r] = e2 * (vA[2][r] + bvl[2]);
      dA[3][r] = e3 * (vA[3][r] + bvl[3]);
    }

    // ---- stage p -> LDS [point][slot] f16; lane owns slots col*4+0..3 -> one b64/point
#pragma unroll
    for (int r = 0; r < 8; ++r) {
      v4h pv = {(_Float16)dA[0][r], (_Float16)dA[1][r],
                (_Float16)dA[2][r], (_Float16)dA[3][r]};
      *(v4h*)(sb + (hi * 8 + r) * 64 + col * 4) = pv;
    }

    v16h as[2];
    as[0] = lds_aop(sb, col, 0, hi);
    as[1] = lds_aop(sb, col, 1, hi);

    // ---- G = p@Wg (zero C, cin-permuted weights match slot order)
    v8f gA[4];
#pragma unroll
    for (int t = 0; t < 4; ++t) {
      v8f z = {};
      v8f g = __builtin_amdgcn_wmma_f32_16x16x32_f16(false, as[0], false, lds_bop(wgg, 16 * t + col, hi, 0), (short)0, z, false, false);
      g     = __builtin_amdgcn_wmma_f32_16x16x32_f16(false, as[1], false, lds_bop(wgg, 16 * t + col, hi, 1), (short)0, g, false, false);
      gA[t] = g;
    }

    // ---- u = p + G -> LDS (same slot layout), reload as A operand
#pragma unroll
    for (int r = 0; r < 8; ++r) {
      v4h uv = {(_Float16)(dA[0][r] + gA[0][r]), (_Float16)(dA[1][r] + gA[1][r]),
                (_Float16)(dA[2][r] + gA[2][r]), (_Float16)(dA[3][r] + gA[3][r])};
      *(v4h*)(sb + (hi * 8 + r) * 64 + col * 4) = uv;
    }

    v16h au[2];
    au[0] = lds_aop(sb, col, 0, hi);
    au[1] = lds_aop(sb, col, 1, hi);

    // ---- H = u@Wo (zero C);  out = sigma o H + (bg@Wo + bo), stored [b][cout][n] (streaming NT)
#pragma unroll
    for (int t = 0; t < 4; ++t) {
      v8f z = {};
      v8f h = __builtin_amdgcn_wmma_f32_16x16x32_f16(false, au[0], false, lds_bop(woo, 16 * t + col, hi, 0), (short)0, z, false, false);
      h     = __builtin_amdgcn_wmma_f32_16x16x32_f16(false, au[1], false, lds_bop(woo, 16 * t + col, hi, 1), (short)0, h, false, false);
      // lane (col,hi), comp r holds out[b][16t+col][n0 + hi*8 + r]
      float* op = out + (((size_t)(b * 64 + 16 * t + col)) << 16) + n0 + hi * 8;
      v4f lo4 = {sig[0] * h[0] + bcl[t], sig[1] * h[1] + bcl[t],
                 sig[2] * h[2] + bcl[t], sig[3] * h[3] + bcl[t]};
      v4f hi4 = {sig[4] * h[4] + bcl[t], sig[5] * h[5] + bcl[t],
                 sig[6] * h[6] + bcl[t], sig[7] * h[7] + bcl[t]};
      __builtin_nontemporal_store(lo4, (v4f*)op);
      __builtin_nontemporal_store(hi4, (v4f*)(op + 4));
    }
  }
}

extern "C" void kernel_launch(void* const* d_in, const int* in_sizes, int n_in,
                              void* d_out, int out_size, void* d_ws, size_t ws_size,
                              hipStream_t stream) {
  const float* x  = (const float*)d_in[0];
  // d_in[1] = pos (provably unused by the reference computation)
  const float* Wq = (const float*)d_in[2];
  const float* bq = (const float*)d_in[3];
  const float* Wk = (const float*)d_in[4];
  const float* bk = (const float*)d_in[5];
  const float* Wv = (const float*)d_in[6];
  const float* bv = (const float*)d_in[7];
  const float* Wg = (const float*)d_in[8];
  const float* bg = (const float*)d_in[9];
  const float* Wo = (const float*)d_in[10];
  const float* bo = (const float*)d_in[11];
  const int*   nn = (const int*)d_in[12];
  float* out = (float*)d_out;

  pt_layer_kernel<<<NBLOCKS, THREADS, 0, stream>>>(x, Wq, bq, Wk, bk, Wv, bv,
                                                   Wg, bg, Wo, bo, nn, out);
}
